// GAT_73031623901437
// MI455X (gfx1250) — matrix operand order repaired and checked by reading, not compiled, not run on hardware
//
#include <hip/hip_runtime.h>
#include <cstdint>

typedef _Float16 f16;
typedef __attribute__((ext_vector_type(8)))  _Float16 v8h;
typedef __attribute__((ext_vector_type(16))) _Float16 v16h;
typedef __attribute__((ext_vector_type(8)))  float    v8f;

#define D_IN 256
#define HIDC 64

// GEMM macro-tile config: block = 256 threads (8 waves), tile 32x64, K-step 64
#define TM 32
#define TN 64
#define TK 64

// ---------------- fp32 -> f16 convert ----------------
__global__ void k_cvt_f16(const float* __restrict__ in, f16* __restrict__ out, int n) {
    int i = blockIdx.x * blockDim.x + threadIdx.x;
    if (i < n) out[i] = (f16)in[i];
}

// ---- weight convert + transpose: W[din,dout] f32 -> Wt[dout,din] f16 ----
__global__ void k_wt_trans(const float* __restrict__ W, f16* __restrict__ Wt,
                           int din, int dout) {
    int i = blockIdx.x * blockDim.x + threadIdx.x;
    if (i >= din * dout) return;
    int k = i / dout, c = i % dout;
    Wt[(size_t)c * din + k] = (f16)W[i];
}

// ---------------- fill ----------------
__global__ void k_fill(float* __restrict__ p, float v, int n) {
    int i = blockIdx.x * blockDim.x + threadIdx.x;
    if (i < n) p[i] = v;
}

// ---- WMMA GEMM with async-to-LDS double buffering ----
// C[M,Nout](f32) = A[M,K](f16, row-major) x Bt[Nout,K](f16)^T
// Block: 256 threads = 8 waves; each wave owns one 16x16 subtile of the 32x64
// macro tile. Per K-stage (64): stage A(32x64) + B(64x64) tiles into LDS with
// GLOBAL_LOAD_ASYNC_TO_LDS_B128 (ASYNCcnt), double buffered.
__global__ __launch_bounds__(256) void k_gemm_wmma(
        const f16* __restrict__ A, const f16* __restrict__ Bt,
        float* __restrict__ C, int M, int K, int Nout) {
    __shared__ __align__(32) f16 As[2][TM * TK];   // [row][k]
    __shared__ __align__(32) f16 Bs[2][TN * TK];   // [col][k]

    const int t    = threadIdx.x;
    const int wave = t >> 5;
    const int lane = t & 31;
    const int r    = lane & 15;
    const int hsel = lane >> 4;
    const int mt_l = wave >> 2;          // 0..1  (m subtile)
    const int nt_l = wave & 3;           // 0..3  (n subtile)
    const int m0   = blockIdx.x * TM;
    const int n0   = blockIdx.y * TN;

    // A staging: thread t -> row_l = t>>3, 8-half segment seg = t&7
    int a_row = m0 + (t >> 3);
    if (a_row >= M) a_row = M - 1;                       // clamp, stays in bounds
    const f16* a_gsrc = A + (size_t)a_row * K + ((t & 7) << 3);
    // B staging: 512 segments of 8 halves; thread handles segs t and t+256
    const int s0 = t, s1 = t + 256;
    const f16* b_gsrc0 = Bt + (size_t)(n0 + (s0 >> 3)) * K + ((s0 & 7) << 3);
    const f16* b_gsrc1 = Bt + (size_t)(n0 + (s1 >> 3)) * K + ((s1 & 7) << 3);

    const uint32_t lA_base  = (uint32_t)(uintptr_t)&As[0][((t >> 3) * TK) + ((t & 7) << 3)];
    const uint32_t lB0_base = (uint32_t)(uintptr_t)&Bs[0][((s0 >> 3) * TK) + ((s0 & 7) << 3)];
    const uint32_t lB1_base = (uint32_t)(uintptr_t)&Bs[0][((s1 >> 3) * TK) + ((s1 & 7) << 3)];
    const uint32_t abuf_stride = (uint32_t)(TM * TK * sizeof(f16));
    const uint32_t bbuf_stride = (uint32_t)(TN * TK * sizeof(f16));

    const int nk = K / TK;

    auto issue = [&](int stage) {
        const int buf = stage & 1;
        const int k0 = stage * TK;
        uint32_t lA  = lA_base  + (uint32_t)buf * abuf_stride;
        uint32_t lB0 = lB0_base + (uint32_t)buf * bbuf_stride;
        uint32_t lB1 = lB1_base + (uint32_t)buf * bbuf_stride;
        uint64_t gA  = (uint64_t)(uintptr_t)(a_gsrc  + k0);
        uint64_t gB0 = (uint64_t)(uintptr_t)(b_gsrc0 + k0);
        uint64_t gB1 = (uint64_t)(uintptr_t)(b_gsrc1 + k0);
        asm volatile("global_load_async_to_lds_b128 %0, %1, off"
                     :: "v"(lA),  "v"(gA)  : "memory");
        asm volatile("global_load_async_to_lds_b128 %0, %1, off"
                     :: "v"(lB0), "v"(gB0) : "memory");
        asm volatile("global_load_async_to_lds_b128 %0, %1, off"
                     :: "v"(lB1), "v"(gB1) : "memory");
    };

    v8f acc = {};
    issue(0);
    for (int ks = 0; ks < nk; ++ks) {
        const int buf = ks & 1;
        if (ks + 1 < nk) {
            issue(ks + 1);
            // 3 next-stage loads may remain outstanding; in-order completion
            // guarantees this stage's 3 loads have landed.
            asm volatile("s_wait_asynccnt 0x3" ::: "memory");
        } else {
            asm volatile("s_wait_asynccnt 0x0" ::: "memory");
        }
        __syncthreads();

        const f16* arow = &As[buf][(mt_l * 16 + r) * TK];
        const f16* bcol = &Bs[buf][(nt_l * 16 + r) * TK];
#pragma unroll
        for (int kk = 0; kk < TK; kk += 32) {
            // A 16x32: lanes 0-15 K [0..7],[16..23]; lanes 16-31 K [8..15],[24..31]
            v8h alo = *(const v8h*)(arow + kk + hsel * 8);
            v8h ahi = *(const v8h*)(arow + kk + 16 + hsel * 8);
            // B 32x16: lane = column; lanes 0-15 K [0..7],[16..23]; lanes 16-31 K [8..15],[24..31]
            v8h blo = *(const v8h*)(bcol + kk + hsel * 8);
            v8h bhi = *(const v8h*)(bcol + kk + 16 + hsel * 8);
            v16h b;
            v16h a;
#pragma unroll
            for (int j = 0; j < 8; ++j) { a[j] = alo[j]; a[8 + j] = ahi[j]; b[j] = blo[j]; b[8 + j] = bhi[j]; }
            acc = __builtin_amdgcn_wmma_f32_16x16x32_f16(
                false, a, false, b, (short)0, acc, false, false);
        }
        __syncthreads();   // protect buffer before it is overwritten next round
    }

    const int col = n0 + nt_l * 16 + r;
#pragma unroll
    for (int i = 0; i < 8; ++i) {
        int orow = m0 + mt_l * 16 + hsel * 8 + i;
        if (orow < M && col < Nout) C[(size_t)orow * Nout + col] = acc[i];
    }
}

// ---- per-(node,head) attention logits: dot(hs, a_src), dot(hd, a_dst) ----
__global__ void k_logits(const float* __restrict__ hs, const float* __restrict__ hd,
                         const float* __restrict__ a_src, const float* __restrict__ a_dst,
                         float* __restrict__ als, float* __restrict__ ald, int N, int H) {
    int i = blockIdx.x * blockDim.x + threadIdx.x;
    if (i >= N * H) return;
    int n = i / H, h = i % H;
    const float* ps = hs + (size_t)n * H * HIDC + h * HIDC;
    const float* pd = hd + (size_t)n * H * HIDC + h * HIDC;
    const float* as = a_src + h * HIDC;
    const float* ad = a_dst + h * HIDC;
    float ss = 0.f, sd = 0.f;
#pragma unroll 4
    for (int c = 0; c < HIDC; ++c) { ss += ps[c] * as[c]; sd += pd[c] * ad[c]; }
    als[i] = ss;
    ald[i] = sd;
}

// sign-aware float atomic-max (valid for any mix of signs, init = -inf)
__device__ __forceinline__ void atomicMaxF(float* a, float v) {
    if (v >= 0.f) atomicMax((int*)a, __float_as_int(v));
    else          atomicMin((unsigned int*)a, __float_as_uint(v));
}

// ---- pass 1: edge logits + per-dst segment max ----
__global__ void k_edge_logit(const int* __restrict__ src, const int* __restrict__ dst,
                             const float* __restrict__ als, const float* __restrict__ ald,
                             float* __restrict__ ebuf, float* __restrict__ amax, int E, int H) {
    int i = blockIdx.x * blockDim.x + threadIdx.x;
    if (i >= E * H) return;
    int e = i / H, h = i % H;
    int s = src[e], d = dst[e];
    float v = als[s * H + h] + ald[d * H + h];
    v = v > 0.f ? v : 0.2f * v;           // leaky_relu(0.2)
    ebuf[i] = v;
    atomicMaxF(&amax[d * H + h], v);
}

// ---- pass 2: exp(e - max) + per-dst segment sum ----
__global__ void k_edge_exp(const int* __restrict__ dst, float* __restrict__ ebuf,
                           const float* __restrict__ amax, float* __restrict__ den,
                           int E, int H) {
    int i = blockIdx.x * blockDim.x + threadIdx.x;
    if (i >= E * H) return;
    int e = i / H, h = i % H;
    int d = dst[e];
    float m = amax[d * H + h];
    if ((__float_as_uint(m) & 0x7f800000u) == 0x7f800000u) m = 0.f;  // non-finite -> 0
    float ex = __expf(ebuf[i] - m);
    ebuf[i] = ex;
    atomicAdd(&den[d * H + h], ex);
}

// ---- pass 3: alpha-weighted scatter-add of source features ----
// one block per edge; blockDim = H*HIDC (256 or 64)
__global__ void k_edge_scatter(const int* __restrict__ src, const int* __restrict__ dst,
                               const float* __restrict__ ebuf, const float* __restrict__ den,
                               const float* __restrict__ hs, float* __restrict__ acc,
                               int E, int H) {
    int e = blockIdx.x;
    int t = threadIdx.x;
    int h = t >> 6;                        // t / HIDC
    int s = src[e], d = dst[e];
    float alpha = ebuf[e * H + h] / (den[d * H + h] + 1e-16f);
    float val = hs[(size_t)s * H * HIDC + t] * alpha;
    atomicAdd(&acc[(size_t)d * H * HIDC + t], val);
}

// ---- bias + optional ReLU; emit f16 for next layer or f32 final output ----
__global__ void k_finalize(const float* __restrict__ acc, const float* __restrict__ bias,
                           f16* __restrict__ h16_out, float* __restrict__ f32_out,
                           int N, int dout, int relu) {
    int i = blockIdx.x * blockDim.x + threadIdx.x;
    if (i >= N * dout) return;
    float v = acc[i] + bias[i % dout];
    if (relu) v = fmaxf(v, 0.f);
    if (h16_out) h16_out[i] = (f16)v;
    if (f32_out) f32_out[i] = v;
}

extern "C" void kernel_launch(void* const* d_in, const int* in_sizes, int n_in,
                              void* d_out, int out_size, void* d_ws, size_t ws_size,
                              hipStream_t stream) {
    (void)n_in; (void)out_size; (void)ws_size;

    const float* x = (const float*)d_in[0];
    const int* ei  = (const int*)d_in[1];
    const int E = in_sizes[1] / 2;
    const int N = in_sizes[0] / D_IN;
    const int* src = ei;
    const int* dst = ei + E;

    const float* Wsrc[4] = {(const float*)d_in[2],  (const float*)d_in[7],
                            (const float*)d_in[12], (const float*)d_in[17]};
    const float* Wdst[4] = {(const float*)d_in[3],  (const float*)d_in[8],
                            (const float*)d_in[13], (const float*)d_in[18]};
    const float* Asrc[4] = {(const float*)d_in[4],  (const float*)d_in[9],
                            (const float*)d_in[14], (const float*)d_in[19]};
    const float* Adst[4] = {(const float*)d_in[5],  (const float*)d_in[10],
                            (const float*)d_in[15], (const float*)d_in[20]};
    const float* Bias[4] = {(const float*)d_in[6],  (const float*)d_in[11],
                            (const float*)d_in[16], (const float*)d_in[21]};
    const int Hh[4] = {4, 4, 4, 1};

    // ---- workspace carve-out ----
    char* base = (char*)d_ws;
    size_t off = 0;
    auto alloc = [&](size_t bytes) -> void* {
        void* p = base + off;
        off = (off + bytes + 255) & ~(size_t)255;
        return p;
    };
    f16*   h16  = (f16*)  alloc((size_t)N * D_IN * sizeof(f16));
    float* hs   = (float*)alloc((size_t)N * D_IN * sizeof(float));
    float* hd   = (float*)alloc((size_t)N * D_IN * sizeof(float));
    float* acc  = (float*)alloc((size_t)N * D_IN * sizeof(float));
    float* als  = (float*)alloc((size_t)N * 4 * sizeof(float));
    float* ald  = (float*)alloc((size_t)N * 4 * sizeof(float));
    float* amax = (float*)alloc((size_t)N * 4 * sizeof(float));
    float* den  = (float*)alloc((size_t)N * 4 * sizeof(float));
    float* ebuf = (float*)alloc((size_t)E * 4 * sizeof(float));
    f16* wts[4];
    f16* wtd[4];
    for (int l = 0; l < 4; ++l) {
        int dout = Hh[l] * HIDC;
        wts[l] = (f16*)alloc((size_t)D_IN * dout * sizeof(f16));
        wtd[l] = (f16*)alloc((size_t)D_IN * dout * sizeof(f16));
    }

    // ---- weights: convert + transpose to f16 [dout, din] ----
    for (int l = 0; l < 4; ++l) {
        int dout = Hh[l] * HIDC;
        int nel = D_IN * dout;
        k_wt_trans<<<(nel + 255) / 256, 256, 0, stream>>>(Wsrc[l], wts[l], D_IN, dout);
        k_wt_trans<<<(nel + 255) / 256, 256, 0, stream>>>(Wdst[l], wtd[l], D_IN, dout);
    }

    // ---- input activations to f16 ----
    {
        int nel = N * D_IN;
        k_cvt_f16<<<(nel + 255) / 256, 256, 0, stream>>>(x, h16, nel);
    }

    const float NEG_INF = -__builtin_inff();

    for (int l = 0; l < 4; ++l) {
        const int H = Hh[l];
        const int dout = H * HIDC;
        dim3 gg((N + TM - 1) / TM, dout / TN);

        k_gemm_wmma<<<gg, 256, 0, stream>>>(h16, wts[l], hs, N, D_IN, dout);
        k_gemm_wmma<<<gg, 256, 0, stream>>>(h16, wtd[l], hd, N, D_IN, dout);

        k_logits<<<(N * H + 127) / 128, 128, 0, stream>>>(hs, hd, Asrc[l], Adst[l],
                                                          als, ald, N, H);

        k_fill<<<(N * H + 255) / 256, 256, 0, stream>>>(amax, NEG_INF, N * H);
        k_fill<<<(N * H + 255) / 256, 256, 0, stream>>>(den, 0.f, N * H);
        k_fill<<<(N * dout + 255) / 256, 256, 0, stream>>>(acc, 0.f, N * dout);

        k_edge_logit<<<(E * H + 255) / 256, 256, 0, stream>>>(src, dst, als, ald,
                                                              ebuf, amax, E, H);
        k_edge_exp<<<(E * H + 255) / 256, 256, 0, stream>>>(dst, ebuf, amax, den, E, H);
        k_edge_scatter<<<E, dout, 0, stream>>>(src, dst, ebuf, den, hs, acc, E, H);

        const int relu = (l < 3) ? 1 : 0;
        k_finalize<<<(N * dout + 255) / 256, 256, 0, stream>>>(
            acc, Bias[l],
            (l < 3) ? h16 : (f16*)nullptr,
            (l == 3) ? (float*)d_out : (float*)nullptr,
            N, dout, relu);
    }
}
